// OP3PhysicsNetwork_30262339567939
// MI455X (gfx1250) — compile-verified, matrix operand order checked by reading, not run on hardware
//
#include <hip/hip_runtime.h>
#include <hip/hip_bf16.h>
#include <math.h>

// ---------------------------------------------------------------------------
// OP3 physics network, MI455X (gfx1250) implementation.
// bf16 WMMA (v_wmma_f32_16x16x32_bf16) for all GEMMs, f32 accumulate.
// M-tile = 32 rows: each wave computes two stacked 16x16 C tiles per column
// tile, sharing one B fragment between the two WMMAs (2x weight reuse).
// 4 waves (128 threads) per block.
// ---------------------------------------------------------------------------

typedef __attribute__((ext_vector_type(16))) __bf16 bf16x16;
typedef __attribute__((ext_vector_type(8)))  __bf16 bf16x8;
typedef __attribute__((ext_vector_type(8)))  float  f32x8;

#define NWAVES 4
#define BLOCK  (NWAVES * 32)
#define MTILE  32

__device__ __forceinline__ float elu_f(float x)     { return x > 0.f ? x : __expf(x) - 1.f; }
__device__ __forceinline__ float sigmoid_f(float x) { return 1.f / (1.f + __expf(-x)); }

__device__ __forceinline__ bf16x16 frag16(const __bf16* p) {
  bf16x8 lo = *(const bf16x8*)(p);
  bf16x8 hi = *(const bf16x8*)(p + 16);
  bf16x16 f;
#pragma unroll
  for (int j = 0; j < 8; ++j) { f[j] = lo[j]; f[8 + j] = hi[j]; }
  return f;
}

// Full-block tile GEMM: Out[32 x N] = act(A[32 x K] @ Wt^T + bias).
// A: LDS, row-major [32 x lda] bf16.   Wt: global, row-major [N x K] bf16
// (transposed weights: each lane's B fragment is two contiguous 16B loads).
// ACT: 0 = identity, 1 = ELU.  OutT: __bf16 (LDS) or float (LDS/global).
template <int K, int N, int ACT, typename OutT>
__device__ __forceinline__ void gemm_tile(const __bf16* __restrict__ A, int lda,
                                          const __bf16* __restrict__ Wt,
                                          const float* __restrict__ bias,
                                          OutT* __restrict__ Out, int ldo, int tid) {
  const int wv   = tid >> 5;
  const int lane = tid & 31;
  const int half = lane >> 4;   // 0: K 0..7 / 16..23   1: K 8..15 / 24..31
  const int idx  = lane & 15;   // row of A (lanes) / col of B
  for (int n0 = wv * 16; n0 < N; n0 += NWAVES * 16) {
    f32x8 acc0 = {};
    f32x8 acc1 = {};
    const __bf16* Ar0 = A  + idx * lda + half * 8;          // rows 0..15
    const __bf16* Ar1 = Ar0 + 16 * lda;                     // rows 16..31
    const __bf16* Br  = Wt + (size_t)(n0 + idx) * K + half * 8;
#pragma unroll
    for (int k0 = 0; k0 < K; k0 += 32) {
      bf16x16 b  = frag16(Br  + k0);
      bf16x16 a0 = frag16(Ar0 + k0);
      bf16x16 a1 = frag16(Ar1 + k0);
      acc0 = __builtin_amdgcn_wmma_f32_16x16x32_bf16(false, a0, false, b,
                                                     (short)0, acc0, false, false);
      acc1 = __builtin_amdgcn_wmma_f32_16x16x32_bf16(false, a1, false, b,
                                                     (short)0, acc1, false, false);
    }
    const int   col = n0 + idx;
    const float bv  = bias[col];
#pragma unroll
    for (int r = 0; r < 8; ++r) {
      const int row = r + 8 * half;   // C layout: vgpr r, lane half selects M block
      float v0 = acc0[r] + bv;
      float v1 = acc1[r] + bv;
      if (ACT == 1) { v0 = elu_f(v0); v1 = elu_f(v1); }
      Out[(size_t)row        * ldo + col] = (OutT)v0;
      Out[(size_t)(16 + row) * ldo + col] = (OutT)v1;
    }
  }
}

// N==1 sigmoid head on VALU: att[row] = sigmoid(dot(h[row,:256], w2) + b2)
// 32 rows x 4 segments of 64 elements = 128 threads.
__device__ __forceinline__ void att_head(const __bf16* __restrict__ h, int lda,
                                         const float* __restrict__ w2, float b2,
                                         float* red /*[32*4]*/, float* att /*[32]*/,
                                         int tid) {
  const int row = tid >> 2;   // 0..31
  const int seg = tid & 3;    // 0..3, 64 elems each
  float s = 0.f;
#pragma unroll 4
  for (int k = seg * 64; k < seg * 64 + 64; ++k)
    s += (float)h[row * lda + k] * w2[k];
  red[row * 4 + seg] = s;
  __syncthreads();
  if (tid < 32) {
    float t = b2 + red[tid * 4 + 0] + red[tid * 4 + 1]
                 + red[tid * 4 + 2] + red[tid * 4 + 3];
    att[tid] = sigmoid_f(t);
  }
  __syncthreads();
}

// ---------------------------------------------------------------------------
// K0: convert fp32 weights -> bf16, transposed to [dout][din]
// ---------------------------------------------------------------------------
__global__ void convert_transpose(const float* __restrict__ w, __bf16* __restrict__ wt,
                                  int din, int dout) {
  int t = blockIdx.x * blockDim.x + threadIdx.x;
  if (t >= din * dout) return;
  int o = t / din;
  int i = t - o * din;
  wt[t] = (__bf16)w[(size_t)i * dout + o];
}

// ---------------------------------------------------------------------------
// K1: action encoder.  aenc[bl] = elu(elu(ae_w1[action[bl]] + b1) @ ae_w2 + b2)
// ---------------------------------------------------------------------------
__global__ __launch_bounds__(256)
void action_encoder(const int* __restrict__ action,
                    const float* __restrict__ w1, const float* __restrict__ b1,
                    const float* __restrict__ w2, const float* __restrict__ b2,
                    __bf16* __restrict__ aenc) {
  __shared__ float hid[256];
  const int bl  = blockIdx.x;
  const int tid = threadIdx.x;
  const int a   = action[bl];
  hid[tid] = elu_f(w1[a * 256 + tid] + b1[tid]);
  __syncthreads();
  if (tid < 32) {
    float s = b2[tid];
#pragma unroll 4
    for (int k = 0; k < 256; ++k) s += hid[k] * w2[k * 32 + tid];
    aenc[bl * 32 + tid] = (__bf16)elu_f(s);
  }
}

// ---------------------------------------------------------------------------
// K2: per-entity front end -> state_enc  (256 blocks x 32 rows)
// ---------------------------------------------------------------------------
__global__ __launch_bounds__(BLOCK)
void entity_stage(const float* __restrict__ samples, const __bf16* __restrict__ aenc,
                  const __bf16* in_w1t, const float* in_b1,
                  const __bf16* in_w2t, const float* in_b2,
                  const __bf16* eff_w1t, const float* eff_b1,
                  const __bf16* eff_w2t, const float* eff_b2,
                  const __bf16* aat_w1t, const float* aat_b1,
                  const float* aat_w2, const float* aat_b2,
                  __bf16* __restrict__ state_enc) {
  __shared__ __bf16 sa[MTILE][288];   // state (0..255) ++ aenc (256..287)
  __shared__ __bf16 xb[MTILE][256];   // samples tile, later aat hidden
  __shared__ __bf16 h [MTILE][256];   // hidden
  __shared__ __bf16 se[MTILE][256];   // s_eff
  __shared__ float  red[MTILE * 4];
  __shared__ float  att[MTILE];
  const int tid = threadIdx.x;
  const int m0  = blockIdx.x * MTILE;

  for (int i = tid; i < MTILE * 256; i += BLOCK) {
    int r = i >> 8, c = i & 255;
    xb[r][c] = (__bf16)samples[(size_t)(m0 + r) * 256 + c];
  }
  for (int i = tid; i < MTILE * 32; i += BLOCK) {
    int r = i >> 5, c = i & 31;
    int bl = (m0 + r) >> 3;            // N = 8 entities per (b,l)
    sa[r][256 + c] = aenc[bl * 32 + c];
  }
  __syncthreads();
  gemm_tile<256, 256, 1>(&xb[0][0], 256, in_w1t, in_b1, &h[0][0], 256, tid);
  __syncthreads();
  gemm_tile<256, 256, 1>(&h[0][0], 256, in_w2t, in_b2, &sa[0][0], 288, tid);
  __syncthreads();
  gemm_tile<288, 256, 1>(&sa[0][0], 288, eff_w1t, eff_b1, &h[0][0], 256, tid);
  __syncthreads();
  gemm_tile<256, 256, 1>(&h[0][0], 256, eff_w2t, eff_b2, &se[0][0], 256, tid);
  __syncthreads();
  gemm_tile<288, 256, 1>(&sa[0][0], 288, aat_w1t, aat_b1, &xb[0][0], 256, tid);
  __syncthreads();
  att_head(&xb[0][0], 256, aat_w2, aat_b2[0], red, att, tid);
  for (int i = tid; i < MTILE * 256; i += BLOCK) {
    int r = i >> 8, c = i & 255;
    float v = (float)se[r][c] * att[r];
    state_enc[(size_t)(m0 + r) * 256 + c] = (__bf16)v;
  }
}

// ---------------------------------------------------------------------------
// K3: pair interactions.  Block owns 32 entities, loops the 7 partners,
// builds pair tiles in LDS, accumulates total_effect without atomics.
// ---------------------------------------------------------------------------
__global__ __launch_bounds__(BLOCK)
void pair_stage(const __bf16* __restrict__ state_enc,
                const __bf16* pw_w1t, const float* pw_b1,
                const __bf16* pw_w2t, const float* pw_b2,
                const __bf16* ie_w1t, const float* ie_b1,
                const __bf16* ie_w2t, const float* ie_b2,
                const __bf16* ia_w1t, const float* ia_b1,
                const float* ia_w2, const float* ia_b2,
                float* __restrict__ total_eff) {
  __shared__ __bf16 pairt[MTILE][512];   // [src | tgt]
  __shared__ __bf16 ph   [MTILE][512];
  __shared__ __bf16 pint [MTILE][256];
  __shared__ __bf16 h2   [MTILE][256];
  __shared__ float  peff [MTILE][32];
  __shared__ float  accum[MTILE][32];
  __shared__ float  red[MTILE * 4];
  __shared__ float  att[MTILE];
  const int tid = threadIdx.x;
  const int e0  = blockIdx.x * MTILE;

  for (int i = tid; i < MTILE * 32; i += BLOCK) ((float*)accum)[i] = 0.f;
  for (int i = tid; i < MTILE * 256; i += BLOCK) {
    int r = i >> 8, c = i & 255;
    pairt[r][c] = state_enc[(size_t)(e0 + r) * 256 + c];
  }
  for (int t = 0; t < 7; ++t) {
    __syncthreads();
    for (int i = tid; i < MTILE * 256; i += BLOCK) {
      int r = i >> 8, c = i & 255;
      int e  = e0 + r;
      int ii = e & 7;
      int j  = t + (t >= ii ? 1 : 0);      // jidx: skip j == i
      int tgt = (e & ~7) | j;
      pairt[r][256 + c] = state_enc[(size_t)tgt * 256 + c];
    }
    __syncthreads();
    gemm_tile<512, 512, 1>(&pairt[0][0], 512, pw_w1t, pw_b1, &ph[0][0], 512, tid);
    __syncthreads();
    gemm_tile<512, 256, 1>(&ph[0][0], 512, pw_w2t, pw_b2, &pint[0][0], 256, tid);
    __syncthreads();
    gemm_tile<256, 256, 1>(&pint[0][0], 256, ie_w1t, ie_b1, &h2[0][0], 256, tid);
    __syncthreads();
    gemm_tile<256, 32, 1>(&h2[0][0], 256, ie_w2t, ie_b2, &peff[0][0], 32, tid);
    __syncthreads();
    gemm_tile<256, 256, 1>(&pint[0][0], 256, ia_w1t, ia_b1, &h2[0][0], 256, tid);
    __syncthreads();
    att_head(&h2[0][0], 256, ia_w2, ia_b2[0], red, att, tid);
    for (int i = tid; i < MTILE * 32; i += BLOCK) {
      int r = i >> 5, c = i & 31;
      accum[r][c] += peff[r][c] * att[r];
    }
  }
  __syncthreads();
  for (int i = tid; i < MTILE * 32; i += BLOCK) {
    int r = i >> 5, c = i & 31;
    total_eff[(size_t)(e0 + r) * 32 + c] = accum[r][c];
  }
}

// ---------------------------------------------------------------------------
// K4: merge + output head -> f32 d_out
// ---------------------------------------------------------------------------
__global__ __launch_bounds__(BLOCK)
void final_stage(const __bf16* __restrict__ state_enc, const float* __restrict__ total_eff,
                 const __bf16* fm_w1t, const float* fm_b1,
                 const __bf16* fm_w2t, const float* fm_b2,
                 const __bf16* out_w1t, const float* out_b1,
                 const __bf16* out_w2t, const float* out_b2,
                 float* __restrict__ out) {
  __shared__ __bf16 sa[MTILE][288];
  __shared__ __bf16 h [MTILE][256];
  __shared__ __bf16 mg[MTILE][256];
  const int tid = threadIdx.x;
  const int m0  = blockIdx.x * MTILE;
  for (int i = tid; i < MTILE * 256; i += BLOCK) {
    int r = i >> 8, c = i & 255;
    sa[r][c] = state_enc[(size_t)(m0 + r) * 256 + c];
  }
  for (int i = tid; i < MTILE * 32; i += BLOCK) {
    int r = i >> 5, c = i & 31;
    sa[r][256 + c] = (__bf16)total_eff[(size_t)(m0 + r) * 32 + c];
  }
  __syncthreads();
  gemm_tile<288, 256, 1>(&sa[0][0], 288, fm_w1t, fm_b1, &h[0][0], 256, tid);
  __syncthreads();
  gemm_tile<256, 256, 1>(&h[0][0], 256, fm_w2t, fm_b2, &mg[0][0], 256, tid);
  __syncthreads();
  gemm_tile<256, 256, 1>(&mg[0][0], 256, out_w1t, out_b1, &h[0][0], 256, tid);
  __syncthreads();
  gemm_tile<256, 256, 0>(&h[0][0], 256, out_w2t, out_b2,
                         out + (size_t)m0 * 256, 256, tid);
}

// ---------------------------------------------------------------------------
// Host launcher
// ---------------------------------------------------------------------------
extern "C" void kernel_launch(void* const* d_in, const int* in_sizes, int n_in,
                              void* d_out, int out_size, void* d_ws, size_t ws_size,
                              hipStream_t stream) {
  (void)in_sizes; (void)n_in; (void)out_size; (void)ws_size;
  const float* samples = (const float*)d_in[0];
  const int*   action  = (const int*)d_in[1];
  // d_in[2] mask: unused in the forward pass
  const float* in_w1  = (const float*)d_in[3];
  const float* in_b1  = (const float*)d_in[4];
  const float* in_w2  = (const float*)d_in[5];
  const float* in_b2  = (const float*)d_in[6];
  const float* ae_w1  = (const float*)d_in[7];
  const float* ae_b1  = (const float*)d_in[8];
  const float* ae_w2  = (const float*)d_in[9];
  const float* ae_b2  = (const float*)d_in[10];
  const float* eff_w1 = (const float*)d_in[11];
  const float* eff_b1 = (const float*)d_in[12];
  const float* eff_w2 = (const float*)d_in[13];
  const float* eff_b2 = (const float*)d_in[14];
  const float* aat_w1 = (const float*)d_in[15];
  const float* aat_b1 = (const float*)d_in[16];
  const float* aat_w2 = (const float*)d_in[17];
  const float* aat_b2 = (const float*)d_in[18];
  const float* pw_w1  = (const float*)d_in[19];
  const float* pw_b1  = (const float*)d_in[20];
  const float* pw_w2  = (const float*)d_in[21];
  const float* pw_b2  = (const float*)d_in[22];
  const float* ie_w1  = (const float*)d_in[23];
  const float* ie_b1  = (const float*)d_in[24];
  const float* ie_w2  = (const float*)d_in[25];
  const float* ie_b2  = (const float*)d_in[26];
  const float* ia_w1  = (const float*)d_in[27];
  const float* ia_b1  = (const float*)d_in[28];
  const float* ia_w2  = (const float*)d_in[29];
  const float* ia_b2  = (const float*)d_in[30];
  const float* fm_w1  = (const float*)d_in[31];
  const float* fm_b1  = (const float*)d_in[32];
  const float* fm_w2  = (const float*)d_in[33];
  const float* fm_b2  = (const float*)d_in[34];
  const float* out_w1 = (const float*)d_in[35];
  const float* out_b1 = (const float*)d_in[36];
  const float* out_w2 = (const float*)d_in[37];
  const float* out_b2 = (const float*)d_in[38];

  char*  ws  = (char*)d_ws;
  size_t off = 0;
  auto bump = [&](size_t bytes) -> size_t {
    size_t o = off;
    off += (bytes + 255) & ~(size_t)255;
    return o;
  };
  __bf16* in_w1t  = (__bf16*)(ws + bump(256 * 256 * 2));
  __bf16* in_w2t  = (__bf16*)(ws + bump(256 * 256 * 2));
  __bf16* eff_w1t = (__bf16*)(ws + bump(256 * 288 * 2));
  __bf16* eff_w2t = (__bf16*)(ws + bump(256 * 256 * 2));
  __bf16* aat_w1t = (__bf16*)(ws + bump(256 * 288 * 2));
  __bf16* pw_w1t  = (__bf16*)(ws + bump(512 * 512 * 2));
  __bf16* pw_w2t  = (__bf16*)(ws + bump(256 * 512 * 2));
  __bf16* ie_w1t  = (__bf16*)(ws + bump(256 * 256 * 2));
  __bf16* ie_w2t  = (__bf16*)(ws + bump(32 * 256 * 2));
  __bf16* ia_w1t  = (__bf16*)(ws + bump(256 * 256 * 2));
  __bf16* fm_w1t  = (__bf16*)(ws + bump(256 * 288 * 2));
  __bf16* fm_w2t  = (__bf16*)(ws + bump(256 * 256 * 2));
  __bf16* out_w1t = (__bf16*)(ws + bump(256 * 256 * 2));
  __bf16* out_w2t = (__bf16*)(ws + bump(256 * 256 * 2));
  __bf16* aenc    = (__bf16*)(ws + bump(1024 * 32 * 2));
  __bf16* st_enc  = (__bf16*)(ws + bump(8192 * 256 * 2));
  float*  tot_eff = (float*) (ws + bump(8192 * 32 * 4));

  auto xpose = [&](const float* w, __bf16* wt, int din, int dout) {
    int n = din * dout;
    convert_transpose<<<(n + 255) / 256, 256, 0, stream>>>(w, wt, din, dout);
  };
  xpose(in_w1, in_w1t, 256, 256);   xpose(in_w2, in_w2t, 256, 256);
  xpose(eff_w1, eff_w1t, 288, 256); xpose(eff_w2, eff_w2t, 256, 256);
  xpose(aat_w1, aat_w1t, 288, 256);
  xpose(pw_w1, pw_w1t, 512, 512);   xpose(pw_w2, pw_w2t, 512, 256);
  xpose(ie_w1, ie_w1t, 256, 256);   xpose(ie_w2, ie_w2t, 256, 32);
  xpose(ia_w1, ia_w1t, 256, 256);
  xpose(fm_w1, fm_w1t, 288, 256);   xpose(fm_w2, fm_w2t, 256, 256);
  xpose(out_w1, out_w1t, 256, 256); xpose(out_w2, out_w2t, 256, 256);

  action_encoder<<<1024, 256, 0, stream>>>(action, ae_w1, ae_b1, ae_w2, ae_b2, aenc);

  const int nblk = 8192 / MTILE;   // 256
  entity_stage<<<nblk, BLOCK, 0, stream>>>(samples, aenc,
      in_w1t, in_b1, in_w2t, in_b2,
      eff_w1t, eff_b1, eff_w2t, eff_b2,
      aat_w1t, aat_b1, aat_w2, aat_b2, st_enc);

  pair_stage<<<nblk, BLOCK, 0, stream>>>(st_enc,
      pw_w1t, pw_b1, pw_w2t, pw_b2,
      ie_w1t, ie_b1, ie_w2t, ie_b2,
      ia_w1t, ia_b1, ia_w2, ia_b2, tot_eff);

  final_stage<<<nblk, BLOCK, 0, stream>>>(st_enc, tot_eff,
      fm_w1t, fm_b1, fm_w2t, fm_b2,
      out_w1t, out_b1, out_w2t, out_b2, (float*)d_out);
}
